// MPModule_26010321945127
// MI455X (gfx1250) — matrix operand order repaired and checked by reading, not compiled, not run on hardware
//
#include <hip/hip_runtime.h>
#include <math.h>

typedef __attribute__((ext_vector_type(16))) _Float16 v16h;
typedef __attribute__((ext_vector_type(8)))  float    v8f;

#define B_    16
#define C_    50
#define H_    256
#define W_    256
#define HS_   64
#define WS_   64
#define K_    450          // 50 ci * 9 taps
#define NKB   15           // 480 / 32 k-blocks
#define TOPK_ 240

// ---- d_out layout (floats), concatenated in reference return order ----
#define N_PATCH (3840 * 1800)          // [3840,50,6,6]
#define P_BIDX  (N_PATCH)
#define P_HIDX  (P_BIDX + 3840)
#define P_WIDX  (P_HIDX + 3840)
#define P_MASK  (P_WIDX + 3840)        // [16,64,64]

// ---- workspace layout (bytes) ----
#define RELU_OFF   ((size_t)0)                        // f16 [16,50,256,256] = 104,857,600 B
#define WFRAG_OFF  ((size_t)104857600)                // f16 4*15*32*16     =      61,440 B
#define SMPOOL_OFF ((size_t)104919040)                // f32 [16,64,64]     =     262,144 B
#define SEL_OFF    ((size_t)105181184)                // i32 [3840]         =      15,360 B

// ---------------------------------------------------------------------------
// 1) Pack conv1 weights into per-lane WMMA B-fragment layout (K x N, f16).
//    B layout (ISA 7.12.2, 16-bit 32x16): half j of lane l -> K = (j<8?j:j+8) + (l>=16?8:0), N = l&15
// ---------------------------------------------------------------------------
__global__ __launch_bounds__(256) void pack_w_kernel(const float* __restrict__ W1,
                                                     _Float16* __restrict__ wfrag) {
    int s = blockIdx.x * 256 + threadIdx.x;
    if (s >= 4 * NKB * 32 * 16) return;
    int coTile = s / (NKB * 512);
    int r      = s % (NKB * 512);
    int kb     = r / 512;
    int lane   = (r % 512) / 16;
    int j      = r % 16;
    int co = coTile * 16 + (lane & 15);
    int k  = kb * 32 + ((j < 8) ? j : j + 8) + ((lane < 16) ? 0 : 8);
    float v = (co < C_ && k < K_) ? W1[co * K_ + k] : 0.f;   // W1[co][ci][kh][kw], k = ci*9+t
    wfrag[s] = (_Float16)v;
}

// ---------------------------------------------------------------------------
// 2) conv1 3x3 (50->50) + bias + ReLU as implicit GEMM via v_wmma_f32_16x16x32_f16.
//    WG = 8 waves: 2 pixel-tiles (16 px) x 4 co-tiles (16 ch). A fragments staged
//    in LDS in the exact per-lane layout: K = (j<8?j:j+8) + (lane>=16?8:0), M = lane&15.
// ---------------------------------------------------------------------------
__global__ __launch_bounds__(256) void conv1_wmma_kernel(const float* __restrict__ x,
                                                         const _Float16* __restrict__ wfrag,
                                                         const float* __restrict__ b1,
                                                         _Float16* __restrict__ relu) {
    __shared__ __attribute__((aligned(32))) _Float16 aFrag[2 * NKB * 32 * 16];  // 30,720 B
    const int tid = threadIdx.x;
    const int n0  = blockIdx.x * 32;                 // 32 pixels per workgroup

    for (int s = tid; s < 2 * NKB * 512; s += 256) { // im2col gather -> A-fragment layout
        int pixTile = s / (NKB * 512);
        int r       = s % (NKB * 512);
        int kb      = r / 512;
        int lane    = (r % 512) / 16;
        int j       = r % 16;
        int pl = pixTile * 16 + (lane & 15);
        int k  = kb * 32 + ((j < 8) ? j : j + 8) + ((lane < 16) ? 0 : 8);
        float v = 0.f;
        if (k < K_) {
            int n  = n0 + pl;
            int b  = n >> 16;
            int h  = (n >> 8) & 255;
            int w  = n & 255;
            int ci = k / 9, t9 = k % 9;
            int hh = h + t9 / 3 - 1, ww = w + t9 % 3 - 1;
            if (hh >= 0 && hh < H_ && ww >= 0 && ww < W_)
                v = x[((b * C_ + ci) << 16) + (hh << 8) + ww];
        }
        aFrag[s] = (_Float16)v;
    }
    __syncthreads();

    const int wv      = tid >> 5;
    const int lane    = tid & 31;
    const int pixTile = wv >> 2;
    const int coTile  = wv & 3;
    const _Float16* aW = aFrag + pixTile * (NKB * 512);
    const _Float16* bW = wfrag + coTile * (NKB * 512);

    v8f acc = {};
#pragma unroll
    for (int kb = 0; kb < NKB; ++kb) {
        v16h a  = *(const v16h*)(aW + kb * 512 + lane * 16);
        v16h bb = *(const v16h*)(bW + kb * 512 + lane * 16);
        acc = __builtin_amdgcn_wmma_f32_16x16x32_f16(false, a, false, bb,
                                                     (short)0, acc, false, false);
    }

    int co = coTile * 16 + (lane & 15);
    if (co < C_) {
        float bias = b1[co];
        int mhi = (lane < 16) ? 0 : 8;               // D: VGPR r -> M = r + mhi, N = lane&15
#pragma unroll
        for (int r = 0; r < 8; ++r) {
            int n  = n0 + pixTile * 16 + r + mhi;
            int b  = n >> 16;
            int hw = n & 65535;
            float v = acc[r] + bias;
            relu[((b * C_ + co) << 16) + hw] = (_Float16)(v > 0.f ? v : 0.f);
        }
    }
}

// ---------------------------------------------------------------------------
// 3) conv2 3x3 (50->2) + softmax(keep ch1) + 4x4 avg pool + LeakyReLU(0.1).
//    One thread per pixel; 16x16 tile per WG; pool via LDS.
// ---------------------------------------------------------------------------
__global__ __launch_bounds__(256) void conv2_pool_kernel(const _Float16* __restrict__ relu,
                                                         const float* __restrict__ W2,
                                                         const float* __restrict__ b2,
                                                         float* __restrict__ smPool) {
    __shared__ float w2s[900];
    __shared__ float smT[256];
    const int tid = threadIdx.x;
    for (int s = tid; s < 900; s += 256) w2s[s] = W2[s];
    __syncthreads();

    int b      = blockIdx.x >> 8;
    int tileId = blockIdx.x & 255;
    int h = ((tileId >> 4) << 4) + (tid >> 4);
    int w = ((tileId & 15) << 4) + (tid & 15);

    float a0 = b2[0], a1 = b2[1];
    for (int ci = 0; ci < C_; ++ci) {
        int base = (b * C_ + ci) << 16;
#pragma unroll
        for (int kh = 0; kh < 3; ++kh) {
            int hh = h + kh - 1;
            if (hh < 0 || hh >= H_) continue;
            int rb = base + (hh << 8);
#pragma unroll
            for (int kw = 0; kw < 3; ++kw) {
                int ww = w + kw - 1;
                if (ww < 0 || ww >= W_) continue;
                float v  = (float)relu[rb + ww];
                int   wi = ci * 9 + kh * 3 + kw;
                a0 = fmaf(v, w2s[wi], a0);
                a1 = fmaf(v, w2s[450 + wi], a1);
            }
        }
    }
    smT[tid] = 1.0f / (1.0f + expf(a0 - a1));        // softmax over 2 ch, channel 1
    __syncthreads();

    if (tid < 16) {                                   // 4x4 avg pool of the 16x16 tile
        int py = tid >> 2, px = tid & 3;
        float s = 0.f;
        for (int r = 0; r < 4; ++r)
            for (int c = 0; c < 4; ++c)
                s += smT[(py * 4 + r) * 16 + px * 4 + c];
        s *= (1.0f / 16.0f);
        if (s < 0.f) s *= 0.1f;                       // LeakyReLU(0.1)
        int hs  = ((tileId >> 4) << 2) + py;
        int ws2 = ((tileId & 15) << 2) + px;
        smPool[(b << 12) + hs * WS_ + ws2] = s;
    }
}

// ---------------------------------------------------------------------------
// 4) Per-position 5x5 mixing: kern_t = Wk[t]*s + bk[t]; final_mask = sum kern*window.
// ---------------------------------------------------------------------------
__global__ __launch_bounds__(256) void mask_mix_kernel(const float* __restrict__ smPool,
                                                       const float* __restrict__ Wk,
                                                       const float* __restrict__ bk,
                                                       float* __restrict__ outMask) {
    int g = blockIdx.x * 256 + threadIdx.x;
    if (g >= B_ * HS_ * WS_) return;
    int b   = g >> 12;
    int pos = g & 4095;
    int i = pos >> 6, j = pos & 63;
    const float* sp = smPool + (b << 12);
    float s   = sp[pos];
    float acc = 0.f;
#pragma unroll
    for (int ki = 0; ki < 5; ++ki) {
        int ii = i + ki - 2;
#pragma unroll
        for (int kj = 0; kj < 5; ++kj) {
            int jj = j + kj - 2;
            float wv = 0.f;
            if (ii >= 0 && ii < HS_ && jj >= 0 && jj < WS_) wv = sp[(ii << 6) + jj];
            int t = ki * 5 + kj;
            acc = fmaf(fmaf(Wk[t], s, bk[t]), wv, acc);
        }
    }
    outMask[g] = acc;
}

// ---------------------------------------------------------------------------
// 5) top-k 240 per batch (bitonic sort 4096 in LDS, value-desc / index-asc ties),
//    then 256-wide bitonic ascending sort of the selected flat indices.
// ---------------------------------------------------------------------------
__global__ __launch_bounds__(256) void topk_kernel(const float* __restrict__ mask,
                                                   float* __restrict__ outF,
                                                   int* __restrict__ selIdx) {
    __shared__ float kv[4096];
    __shared__ int   kix[4096];
    __shared__ int   sel[256];
    const int tid = threadIdx.x;
    const int b   = blockIdx.x;
    for (int i = tid; i < 4096; i += 256) { kv[i] = mask[(b << 12) + i]; kix[i] = i; }

    for (unsigned k = 2; k <= 4096; k <<= 1)
        for (unsigned j = k >> 1; j > 0; j >>= 1) {
            __syncthreads();
            for (unsigned t = tid; t < 4096; t += 256) {
                unsigned x = t ^ j;
                if (x > t) {
                    float va = kv[t], vb = kv[x];
                    int   ia = kix[t], ib = kix[x];
                    bool bef = (va > vb) || (va == vb && ia < ib);
                    bool up  = ((t & k) == 0);
                    if (up ? !bef : bef) { kv[t] = vb; kv[x] = va; kix[t] = ib; kix[x] = ia; }
                }
            }
        }
    __syncthreads();

    sel[tid] = (tid < TOPK_) ? kix[tid] : 0x7FFFFFFF;
    for (unsigned k = 2; k <= 256; k <<= 1)
        for (unsigned j = k >> 1; j > 0; j >>= 1) {
            __syncthreads();
            unsigned x = (unsigned)tid ^ j;
            if (x > (unsigned)tid) {
                int a = sel[tid], c = sel[x];
                bool up = (((unsigned)tid & k) == 0);
                if (up ? (a > c) : (a < c)) { sel[tid] = c; sel[x] = a; }
            }
        }
    __syncthreads();

    if (tid < TOPK_) {
        int idx = sel[tid];
        int g   = b * TOPK_ + tid;
        outF[P_BIDX + g] = (float)b;
        outF[P_HIDX + g] = (float)(idx >> 6);
        outF[P_WIDX + g] = (float)(idx & 63);
        selIdx[g] = idx;
    }
}

// ---------------------------------------------------------------------------
// 6) Gather 6x6x50 patches (pad 1, stride 4) from out_lr at selected positions.
// ---------------------------------------------------------------------------
__global__ __launch_bounds__(256) void patch_kernel(const float* __restrict__ x,
                                                    const int* __restrict__ selIdx,
                                                    float* __restrict__ outF) {
    const int g   = blockIdx.x;
    const int b   = g / TOPK_;
    const int idx = selIdx[g];
    const int h0  = (idx >> 6) * 4 - 1;
    const int w0  = (idx & 63) * 4 - 1;
    for (int e = threadIdx.x; e < 1800; e += 256) {
        int c  = e / 36;
        int rr = (e % 36) / 6;
        int cc = e % 6;
        int hh = h0 + rr, ww = w0 + cc;
        float v = 0.f;
        if (hh >= 0 && hh < H_ && ww >= 0 && ww < W_)
            v = x[((b * C_ + c) << 16) + (hh << 8) + ww];
        outF[g * 1800 + e] = v;
    }
}

// ---------------------------------------------------------------------------
extern "C" void kernel_launch(void* const* d_in, const int* in_sizes, int n_in,
                              void* d_out, int out_size, void* d_ws, size_t ws_size,
                              hipStream_t stream) {
    (void)in_sizes; (void)n_in; (void)out_size; (void)ws_size;
    const float* out_lr = (const float*)d_in[0];
    const float* W1     = (const float*)d_in[1];
    const float* b1     = (const float*)d_in[2];
    const float* W2     = (const float*)d_in[3];
    const float* b2     = (const float*)d_in[4];
    const float* Wk     = (const float*)d_in[5];
    const float* bk     = (const float*)d_in[6];
    float* outF = (float*)d_out;
    char*  ws   = (char*)d_ws;
    _Float16* relu   = (_Float16*)(ws + RELU_OFF);
    _Float16* wfrag  = (_Float16*)(ws + WFRAG_OFF);
    float*    smPool = (float*)(ws + SMPOOL_OFF);
    int*      selIdx = (int*)(ws + SEL_OFF);

    pack_w_kernel   <<<dim3(120),   dim3(256), 0, stream>>>(W1, wfrag);
    conv1_wmma_kernel<<<dim3(32768), dim3(256), 0, stream>>>(out_lr, wfrag, b1, relu);
    conv2_pool_kernel<<<dim3(4096),  dim3(256), 0, stream>>>(relu, W2, b2, smPool);
    mask_mix_kernel <<<dim3(256),   dim3(256), 0, stream>>>(smPool, Wk, bk, outF + P_MASK);
    topk_kernel     <<<dim3(16),    dim3(256), 0, stream>>>(outF + P_MASK, outF, selIdx);
    patch_kernel    <<<dim3(3840),  dim3(256), 0, stream>>>(out_lr, selIdx, outF);
}
